// EquivariantGraphEncoder_7902739824976
// MI455X (gfx1250) — matrix operand order repaired and checked by reading, not compiled, not run on hardware
//
#include <hip/hip_runtime.h>

#define N_WAVES 8          // waves per block
#define TILE_E 16          // edges per wave
#define TSTRIDE 40         // f16 tile row stride (conflict-free, 16B-aligned rows)
#define W0STRIDE 72        // e_w0 f16 row stride (cols 0..63)

typedef _Float16 v16h __attribute__((ext_vector_type(16)));
typedef _Float16 h8v  __attribute__((ext_vector_type(8)));
typedef float    v8f  __attribute__((ext_vector_type(8)));

// Fast SiLU: v * sigmoid(v) with v_rcp_f32 instead of IEEE divide.
// Inputs already passed through f16 matmuls, so 1-ulp rcp is lossless in context.
__device__ __forceinline__ float silu_f(float v) {
    return v * __builtin_amdgcn_rcpf(1.0f + __expf(-v));
}

// A-fragment (16x32 f16, M rows striped across VGPRs within a lane).
// lanes 0-15: M=lane, K 0..7 (v0-3) and 16..23 (v4-7); lanes 16-31: M=lane-16, K 8..15 / 24..31.
__device__ __forceinline__ v16h load_frag_a(const _Float16* __restrict__ t, int lane) {
    const int m = lane & 15, g = lane >> 4;
    v16h a;
    *(h8v*)&a       = *(const h8v*)(t + m * TSTRIDE + g * 8);
    *((h8v*)&a + 1) = *(const h8v*)(t + m * TSTRIDE + 16 + g * 8);
    return a;
}

// B-fragment (32x16 f16): lane holds column N = n0 + lane%16, K = k0 + 8*(lane/16) .. +7 and k0 + 16 + 8*(lane/16) .. +7.
// W stored row-major (out x in), so that's 8 consecutive halves of row N at kb and 8 more at kb + 16.
__device__ __forceinline__ v16h load_frag_b(const _Float16* __restrict__ wgt, int stride,
                                            int n0, int k0, int lane) {
    const int n = n0 + (lane & 15);
    const int kb = k0 + (lane >> 4) * 8;
    const _Float16* p = wgt + n * stride + kb;
    v16h b;
    *(h8v*)&b       = *(const h8v*)p;
    *((h8v*)&b + 1) = *(const h8v*)(p + 16);
    return b;
}

// ---------------------------------------------------------------------------
// Edge kernel: per-layer message MLP + coordinate weights, scatter via atomics.
// ---------------------------------------------------------------------------
__global__ __launch_bounds__(256) void egnn_edge_kernel(
    const float* __restrict__ h, const float* __restrict__ x,
    const long long* __restrict__ eidx, const float* __restrict__ eattr,
    const float* __restrict__ e_w0, const float* __restrict__ e_b0,
    const float* __restrict__ e_w1, const float* __restrict__ e_b1,
    const float* __restrict__ c_w0, const float* __restrict__ c_b0,
    const float* __restrict__ c_w1,
    float* __restrict__ magg, float* __restrict__ agg, long long nE) {

    __shared__ __align__(16) _Float16 sW0[32 * W0STRIDE];   // e_w0 cols 0..63 (f16)
    __shared__ __align__(16) _Float16 sW1[32 * TSTRIDE];    // e_w1
    __shared__ __align__(16) _Float16 sCW0[32 * TSTRIDE];   // c_w0
    __shared__ float sW0x[32 * 5];                          // e_w0 cols 64..68 (f32)
    __shared__ float sB0[32], sB1[32], sCB0[32], sCW1[32];
    __shared__ __align__(16) _Float16 tHr[N_WAVES][TILE_E * TSTRIDE];
    __shared__ __align__(16) _Float16 tHc[N_WAVES][TILE_E * TSTRIDE];
    __shared__ __align__(16) _Float16 tM [N_WAVES][TILE_E * TSTRIDE];
    __shared__ float sCD[N_WAVES][TILE_E][4];   // coord_diff xyz + radial
    __shared__ float sEA[N_WAVES][TILE_E][4];   // edge_attr
    __shared__ int   sRow[N_WAVES][TILE_E];
    __shared__ float sS [N_WAVES][TILE_E];

    const int tid = threadIdx.x;
    // ---- stage & convert weights (once per block) ----
    for (int idx = tid; idx < 32 * 64; idx += 256) {
        const int n = idx >> 6, k = idx & 63;
        sW0[n * W0STRIDE + k] = (_Float16)e_w0[n * 69 + k];
    }
    for (int idx = tid; idx < 32 * 32; idx += 256) {
        const int n = idx >> 5, k = idx & 31;
        sW1 [n * TSTRIDE + k] = (_Float16)e_w1[n * 32 + k];
        sCW0[n * TSTRIDE + k] = (_Float16)c_w0[n * 32 + k];
    }
    for (int idx = tid; idx < 32 * 5; idx += 256) {
        const int n = idx / 5, k = idx % 5;
        sW0x[idx] = e_w0[n * 69 + 64 + k];
    }
    if (tid < 32) {
        sB0[tid] = e_b0[tid]; sB1[tid] = e_b1[tid];
        sCB0[tid] = c_b0[tid]; sCW1[tid] = c_w1[tid];
    }
    __syncthreads();

    const int w = tid >> 5, l = tid & 31;

    // ---- phase 1: gather 16 edges per wave (2 lanes per edge) ----
    {
        const int i = l >> 1, half = l & 1;
        const long long e = (long long)blockIdx.x * (N_WAVES * TILE_E) + w * TILE_E + i;
        const bool valid = (e < nE);
        const long long ec = valid ? e : (nE - 1);
        const long long row = eidx[ec];
        const long long col = eidx[nE + ec];

        _Float16 tmp[16] __attribute__((aligned(16)));
        const float* hr = h + row * 32 + half * 16;
        #pragma unroll
        for (int k = 0; k < 16; k += 4) {
            const float4 v = *(const float4*)(hr + k);
            tmp[k] = (_Float16)v.x; tmp[k+1] = (_Float16)v.y;
            tmp[k+2] = (_Float16)v.z; tmp[k+3] = (_Float16)v.w;
        }
        _Float16* dr = &tHr[w][i * TSTRIDE + half * 16];
        *(h8v*)dr = *(const h8v*)tmp; *(h8v*)(dr + 8) = *(const h8v*)(tmp + 8);

        const float* hc = h + col * 32 + half * 16;
        #pragma unroll
        for (int k = 0; k < 16; k += 4) {
            const float4 v = *(const float4*)(hc + k);
            tmp[k] = (_Float16)v.x; tmp[k+1] = (_Float16)v.y;
            tmp[k+2] = (_Float16)v.z; tmp[k+3] = (_Float16)v.w;
        }
        _Float16* dc = &tHc[w][i * TSTRIDE + half * 16];
        *(h8v*)dc = *(const h8v*)tmp; *(h8v*)(dc + 8) = *(const h8v*)(tmp + 8);

        if (half == 0) {
            const float cd0 = x[row * 3 + 0] - x[col * 3 + 0];
            const float cd1 = x[row * 3 + 1] - x[col * 3 + 1];
            const float cd2 = x[row * 3 + 2] - x[col * 3 + 2];
            sCD[w][i][0] = cd0; sCD[w][i][1] = cd1; sCD[w][i][2] = cd2;
            sCD[w][i][3] = cd0 * cd0 + cd1 * cd1 + cd2 * cd2;   // radial
            const float4 ea = *(const float4*)(eattr + ec * 4);
            sEA[w][i][0] = ea.x; sEA[w][i][1] = ea.y; sEA[w][i][2] = ea.z; sEA[w][i][3] = ea.w;
            sRow[w][i] = valid ? (int)row : -1;
        }
    }
    __syncthreads();

    // ---- phase 2: m1 = silu([h_r | h_c | radial | ea] @ e_w0^T + e_b0) ----
    const v16h a1 = load_frag_a(&tHr[w][0], l);
    const v16h a2 = load_frag_a(&tHc[w][0], l);
    const int nlo = l & 15, mhi = (l >> 4) << 3;

    #pragma unroll
    for (int nh = 0; nh < 2; ++nh) {
        const int n = nh * 16 + nlo;
        const float b0v = sB0[n];
        const float w64 = sW0x[n * 5 + 0], w65 = sW0x[n * 5 + 1], w66 = sW0x[n * 5 + 2];
        const float w67 = sW0x[n * 5 + 3], w68 = sW0x[n * 5 + 4];
        v8f acc;
        #pragma unroll
        for (int r = 0; r < 8; ++r) {
            const int M = r + mhi;
            acc[r] = b0v + sCD[w][M][3] * w64
                   + sEA[w][M][0] * w65 + sEA[w][M][1] * w66
                   + sEA[w][M][2] * w67 + sEA[w][M][3] * w68;
        }
        acc = __builtin_amdgcn_wmma_f32_16x16x32_f16(false, a1, false,
                load_frag_b(sW0, W0STRIDE, nh * 16, 0, l), (short)0, acc, false, false);
        acc = __builtin_amdgcn_wmma_f32_16x16x32_f16(false, a2, false,
                load_frag_b(sW0, W0STRIDE, nh * 16, 32, l), (short)0, acc, false, false);
        #pragma unroll
        for (int r = 0; r < 8; ++r) {
            const int M = r + mhi;
            tM[w][M * TSTRIDE + n] = (_Float16)silu_f(acc[r]);
        }
    }
    __syncthreads();

    // ---- phase 3: m = silu(m1 @ e_w1^T + e_b1); keep in regs + store f16 for next A ----
    const v16h am = load_frag_a(&tM[w][0], l);
    float mval[2][8];
    #pragma unroll
    for (int nh = 0; nh < 2; ++nh) {
        const int n = nh * 16 + nlo;
        const float bv = sB1[n];
        v8f acc;
        #pragma unroll
        for (int r = 0; r < 8; ++r) acc[r] = bv;
        acc = __builtin_amdgcn_wmma_f32_16x16x32_f16(false, am, false,
                load_frag_b(sW1, TSTRIDE, nh * 16, 0, l), (short)0, acc, false, false);
        #pragma unroll
        for (int r = 0; r < 8; ++r) {
            const int M = r + mhi;
            const float v = silu_f(acc[r]);
            mval[nh][r] = v;
            tHr[w][M * TSTRIDE + n] = (_Float16)v;   // reuse tHr as m tile
        }
    }
    __syncthreads();

    // ---- phase 4: c = silu(m @ c_w0^T + c_b0); s = c . c_w1 (shuffle reduction) ----
    const v16h ac = load_frag_a(&tHr[w][0], l);
    float cval[2][8];
    #pragma unroll
    for (int nh = 0; nh < 2; ++nh) {
        const int n = nh * 16 + nlo;
        const float bv = sCB0[n];
        v8f acc;
        #pragma unroll
        for (int r = 0; r < 8; ++r) acc[r] = bv;
        acc = __builtin_amdgcn_wmma_f32_16x16x32_f16(false, ac, false,
                load_frag_b(sCW0, TSTRIDE, nh * 16, 0, l), (short)0, acc, false, false);
        #pragma unroll
        for (int r = 0; r < 8; ++r) cval[nh][r] = silu_f(acc[r]);
    }
    {
        const float wa = sCW1[nlo], wb = sCW1[16 + nlo];
        #pragma unroll
        for (int r = 0; r < 8; ++r) {
            float t = cval[0][r] * wa + cval[1][r] * wb;
            t += __shfl_xor(t, 1); t += __shfl_xor(t, 2);
            t += __shfl_xor(t, 4); t += __shfl_xor(t, 8);   // stays inside 16-lane group
            if (nlo == 0) sS[w][r + mhi] = t;
        }
    }
    __syncthreads();

    // ---- phase 5: scatter (f32 atomics into L2-resident buffers) ----
    #pragma unroll
    for (int nh = 0; nh < 2; ++nh) {
        const int n = nh * 16 + nlo;
        #pragma unroll
        for (int r = 0; r < 8; ++r) {
            const int M = r + mhi;
            const int rw = sRow[w][M];
            if (rw >= 0) atomicAdd(&magg[(size_t)rw * 32 + n], mval[nh][r]);
        }
    }
    if (l < TILE_E) {
        const int rw = sRow[w][l];
        if (rw >= 0) {
            const float s = sS[w][l];
            atomicAdd(&agg[(size_t)rw * 3 + 0], sCD[w][l][0] * s);
            atomicAdd(&agg[(size_t)rw * 3 + 1], sCD[w][l][1] * s);
            atomicAdd(&agg[(size_t)rw * 3 + 2], sCD[w][l][2] * s);
        }
    }
}

// ---------------------------------------------------------------------------
// Node kernel: x += agg/max(cnt,1); h += n_w1 @ silu(n_w0 @ [h|magg] + b0) + b1
// ---------------------------------------------------------------------------
__global__ __launch_bounds__(256) void egnn_node_kernel(
    float* __restrict__ h, float* __restrict__ x,
    const float* __restrict__ magg, const float* __restrict__ agg,
    const float* __restrict__ cnt,
    const float* __restrict__ n_w0, const float* __restrict__ n_b0,
    const float* __restrict__ n_w1, const float* __restrict__ n_b1, int N) {
    __shared__ float sW0[32 * 64], sW1[32 * 32], sB0[32], sB1[32];
    const int tid = threadIdx.x;
    for (int idx = tid; idx < 32 * 64; idx += 256) sW0[idx] = n_w0[idx];
    for (int idx = tid; idx < 32 * 32; idx += 256) sW1[idx] = n_w1[idx];
    if (tid < 32) { sB0[tid] = n_b0[tid]; sB1[tid] = n_b1[tid]; }
    __syncthreads();

    const int i = blockIdx.x * 256 + tid;
    if (i >= N) return;

    const float c = fmaxf(cnt[i], 1.0f);
    #pragma unroll
    for (int d = 0; d < 3; ++d) x[(size_t)i * 3 + d] += agg[(size_t)i * 3 + d] / c;

    float hv[32], mg[32];
    #pragma unroll
    for (int k = 0; k < 32; ++k) { hv[k] = h[(size_t)i * 32 + k]; mg[k] = magg[(size_t)i * 32 + k]; }
    float nm[32];
    #pragma unroll 4
    for (int o = 0; o < 32; ++o) {
        float acc = sB0[o];
        #pragma unroll
        for (int k = 0; k < 32; ++k) acc += hv[k] * sW0[o * 64 + k];
        #pragma unroll
        for (int k = 0; k < 32; ++k) acc += mg[k] * sW0[o * 64 + 32 + k];
        nm[o] = silu_f(acc);
    }
    #pragma unroll 4
    for (int o = 0; o < 32; ++o) {
        float acc = sB1[o];
        #pragma unroll
        for (int k = 0; k < 32; ++k) acc += nm[k] * sW1[o * 32 + k];
        h[(size_t)i * 32 + o] = hv[o] + acc;
    }
}

// ---------------------------------------------------------------------------
__global__ __launch_bounds__(256) void egnn_embed_kernel(
    const float* __restrict__ h_in, const float* __restrict__ x_in,
    const float* __restrict__ wgt, const float* __restrict__ bias,
    float* __restrict__ h, float* __restrict__ x, int N) {
    __shared__ float sW[32 * 13], sB[32];
    const int tid = threadIdx.x;
    for (int idx = tid; idx < 32 * 13; idx += 256) sW[idx] = wgt[idx];
    if (tid < 32) sB[tid] = bias[tid];
    __syncthreads();
    const int i = blockIdx.x * 256 + tid;
    if (i >= N) return;
    float in[13];
    #pragma unroll
    for (int k = 0; k < 13; ++k) in[k] = h_in[(size_t)i * 13 + k];
    #pragma unroll 4
    for (int o = 0; o < 32; ++o) {
        float acc = sB[o];
        #pragma unroll
        for (int k = 0; k < 13; ++k) acc += in[k] * sW[o * 13 + k];
        h[(size_t)i * 32 + o] = acc;
    }
    #pragma unroll
    for (int d = 0; d < 3; ++d) x[(size_t)i * 3 + d] = x_in[(size_t)i * 3 + d];
}

__global__ __launch_bounds__(256) void egnn_out_kernel(
    float* __restrict__ h, const float* __restrict__ wgt,
    const float* __restrict__ bias, int N) {
    __shared__ float sW[32 * 32], sB[32];
    const int tid = threadIdx.x;
    for (int idx = tid; idx < 32 * 32; idx += 256) sW[idx] = wgt[idx];
    if (tid < 32) sB[tid] = bias[tid];
    __syncthreads();
    const int i = blockIdx.x * 256 + tid;
    if (i >= N) return;
    float hv[32];
    #pragma unroll
    for (int k = 0; k < 32; ++k) hv[k] = h[(size_t)i * 32 + k];
    #pragma unroll 4
    for (int o = 0; o < 32; ++o) {
        float acc = sB[o];
        #pragma unroll
        for (int k = 0; k < 32; ++k) acc += hv[k] * sW[o * 32 + k];
        h[(size_t)i * 32 + o] = acc;   // in-place, per-node safe
    }
}

__global__ __launch_bounds__(256) void egnn_count_kernel(
    const long long* __restrict__ eidx, float* __restrict__ cnt, long long nE) {
    const long long e = (long long)blockIdx.x * 256 + threadIdx.x;
    if (e < nE) atomicAdd(&cnt[eidx[e]], 1.0f);
}

// ---------------------------------------------------------------------------
extern "C" void kernel_launch(void* const* d_in, const int* in_sizes, int n_in,
                              void* d_out, int out_size, void* d_ws, size_t ws_size,
                              hipStream_t stream) {
    // params flattened in insertion order: emb_in_w, emb_in_b, emb_out_w, emb_out_b,
    // then per layer: e_w0,e_b0,e_w1,e_b1,n_w0,n_b0,n_w1,n_b1,c_w0,c_b0,c_w1;
    // then h, x, edge_index (int64), edge_attr.
    const float* emb_in_w  = (const float*)d_in[0];
    const float* emb_in_b  = (const float*)d_in[1];
    const float* emb_out_w = (const float*)d_in[2];
    const float* emb_out_b = (const float*)d_in[3];
    const float* h_in   = (const float*)d_in[37];
    const float* x_in   = (const float*)d_in[38];
    const long long* ei = (const long long*)d_in[39];
    const float* eattr  = (const float*)d_in[40];

    const int N = in_sizes[37] / 13;
    const long long nE = (long long)in_sizes[40] / 4;

    float* h = (float*)d_out;                    // N x 32 (output region, used as working h)
    float* x = h + (size_t)N * 32;               // N x 3  (output region, used as working x)
    float* magg = (float*)d_ws;                  // N x 32
    float* agg  = magg + (size_t)N * 32;         // N x 3
    float* cnt  = agg + (size_t)N * 3;           // N

    const int nodeBlocks = (N + 255) / 256;
    const int edgeBlocks = (int)((nE + (N_WAVES * TILE_E) - 1) / (N_WAVES * TILE_E));
    const int cntBlocks  = (int)((nE + 255) / 256);

    hipMemsetAsync(cnt, 0, (size_t)N * sizeof(float), stream);
    egnn_count_kernel<<<cntBlocks, 256, 0, stream>>>(ei, cnt, nE);
    egnn_embed_kernel<<<nodeBlocks, 256, 0, stream>>>(h_in, x_in, emb_in_w, emb_in_b, h, x, N);

    for (int lyr = 0; lyr < 3; ++lyr) {
        const int b = 4 + 11 * lyr;
        const float* e_w0 = (const float*)d_in[b + 0];
        const float* e_b0 = (const float*)d_in[b + 1];
        const float* e_w1 = (const float*)d_in[b + 2];
        const float* e_b1 = (const float*)d_in[b + 3];
        const float* n_w0 = (const float*)d_in[b + 4];
        const float* n_b0 = (const float*)d_in[b + 5];
        const float* n_w1 = (const float*)d_in[b + 6];
        const float* n_b1 = (const float*)d_in[b + 7];
        const float* c_w0 = (const float*)d_in[b + 8];
        const float* c_b0 = (const float*)d_in[b + 9];
        const float* c_w1 = (const float*)d_in[b + 10];

        hipMemsetAsync(magg, 0, (size_t)N * 32 * sizeof(float), stream);
        hipMemsetAsync(agg, 0, (size_t)N * 3 * sizeof(float), stream);
        egnn_edge_kernel<<<edgeBlocks, 256, 0, stream>>>(
            h, x, ei, eattr, e_w0, e_b0, e_w1, e_b1, c_w0, c_b0, c_w1, magg, agg, nE);
        egnn_node_kernel<<<nodeBlocks, 256, 0, stream>>>(
            h, x, magg, agg, cnt, n_w0, n_b0, n_w1, n_b1, N);
    }
    egnn_out_kernel<<<nodeBlocks, 256, 0, stream>>>(h, emb_out_w, emb_out_b, N);
}
